// SDE_35218731827222
// MI455X (gfx1250) — matrix operand order repaired and checked by reading, not compiled, not run on hardware
//
#include <hip/hip_runtime.h>

#define VOCAB  30000
#define DIM    768
#define LATENT 5000
#define LATPAD 5120
#define NROWS  (16 * 1024)
#define CCH    8

typedef __attribute__((ext_vector_type(16))) __bf16 v16bf;
typedef __attribute__((ext_vector_type(8)))  float  v8f;

union Frag {
    v16bf v;
    uint4 q[2];
};

__device__ __forceinline__ unsigned short f2bf(float f) {
    union { float f; unsigned int u; } x;
    x.f = f;
    unsigned int r = x.u + 0x7FFFu + ((x.u >> 16) & 1u);  // round-to-nearest-even
    return (unsigned short)(r >> 16);
}

__device__ __forceinline__ v8f zero8() {
    v8f z;
#pragma unroll
    for (int i = 0; i < 8; ++i) z[i] = 0.0f;
    return z;
}

// ---------------------------------------------------------------------------
// Kernel 1: lang_emb = tanh(masked-mean of char embeddings)
//   writes f32 into d_out (later read-modify-written by the fused kernel)
//   and a bf16 copy for the WMMA GEMMs.
// ---------------------------------------------------------------------------
__global__ __launch_bounds__(256) void lang_kernel(
    const int* __restrict__ x, const float* __restrict__ emb,
    float* __restrict__ out, unsigned short* __restrict__ langbf) {
    const int row = blockIdx.x;
    const int tid = threadIdx.x;

    int ch[CCH];
    int cnt = 0;
#pragma unroll
    for (int c = 0; c < CCH; ++c) {
        ch[c] = x[row * CCH + c];
        cnt += (ch[c] != 0);
    }
    const float inv = 1.0f / (float)(cnt > 0 ? cnt : 1);

#pragma unroll
    for (int j = 0; j < DIM / 256; ++j) {
        const int d = tid + j * 256;
        float s = 0.0f;
#pragma unroll
        for (int c = 0; c < CCH; ++c) {
            const float m = (ch[c] != 0) ? 1.0f : 0.0f;
            s += m * emb[ch[c] * DIM + d];  // row 0 read is harmless when masked
        }
        const float t = tanhf(s * inv);
        out[row * DIM + d]    = t;
        langbf[row * DIM + d] = f2bf(t);
    }
}

// ---------------------------------------------------------------------------
// Kernel 2: latent_mat f32 -> bf16 (padded to 5120 rows with zeros), plus an
// LDS-tiled transpose latT[d][k] so GEMM2 B-fragments are contiguous loads.
// ---------------------------------------------------------------------------
__global__ __launch_bounds__(256) void prep_latent(
    const float* __restrict__ lm,
    unsigned short* __restrict__ latbf, unsigned short* __restrict__ latT) {
    __shared__ unsigned short tile[32][33];
    const int k0 = blockIdx.x * 32, d0 = blockIdx.y * 32;
    const int tx = threadIdx.x, ty = threadIdx.y;
#pragma unroll
    for (int i = 0; i < 32; i += 8) {
        const int k = k0 + ty + i;
        const float v = (k < LATENT) ? lm[k * DIM + d0 + tx] : 0.0f;
        const unsigned short b = f2bf(v);
        latbf[k * DIM + d0 + tx] = b;
        tile[ty + i][tx] = b;
    }
    __syncthreads();
#pragma unroll
    for (int i = 0; i < 32; i += 8) {
        const int d = d0 + ty + i;
        latT[d * LATPAD + k0 + tx] = tile[tx][ty + i];
    }
}

// ---------------------------------------------------------------------------
// Kernel 3: fused  scores -> softmax -> P@V  (flash-attention style),
// plus lang_emb add and special-token override.
// Block = 16 query rows, 8 wave32s. Each wave: one 16x16 score tile per
// 128-latent step (GEMM1), then a private 16x96 output slice (GEMM2).
// ---------------------------------------------------------------------------
__global__ __launch_bounds__(256) void sde_attn(
    const float* __restrict__ emb, const int* __restrict__ x,
    const unsigned short* __restrict__ langbf,
    const unsigned short* __restrict__ latbf,
    const unsigned short* __restrict__ latT,
    float* __restrict__ out) {
    __shared__ __align__(16) unsigned short Qs[16 * DIM];  // 24 KB
    __shared__ __align__(16) unsigned short Ps[16 * 128];  //  4 KB
    __shared__ float sumRow[16];

    const int tid  = threadIdx.x;
    const int w    = tid >> 5;    // wave id 0..7
    const int lane = tid & 31;
    const int half = lane >> 4;   // 0: rows 0-7 / K-lo, 1: rows 8-15 / K-hi
    const int ln   = lane & 15;
    const int row0 = blockIdx.x * 16;

    // stage Q tile (16 x 768 bf16) into LDS, zero row-sum accumulators
    {
        const uint4* qg = (const uint4*)(langbf + (size_t)row0 * DIM);
        uint4* qs = (uint4*)Qs;
        for (int i = tid; i < 16 * DIM / 8; i += 256) qs[i] = qg[i];
    }
    if (tid < 16) sumRow[tid] = 0.0f;
    __syncthreads();

    v8f acc[6];
#pragma unroll
    for (int nt = 0; nt < 6; ++nt) acc[nt] = zero8();
    float rsum[8];
#pragma unroll
    for (int r = 0; r < 8; ++r) rsum[r] = 0.0f;

    const int NKB = LATPAD / 128;  // 40 outer steps
    for (int kb = 0; kb < NKB; ++kb) {
        const int lat0 = kb * 128 + w * 16;
        const int latn = lat0 + ln;  // this lane's latent column

        const unsigned short* brow = latbf + (size_t)latn * DIM;
        if (kb + 1 < NKB)
            __builtin_prefetch(brow + 128 * DIM, 0, 3);  // next latent block

        // ---- GEMM1: S(16x16) = Q(16x768) x latent^T, 24 k-steps of 32 ----
        v8f s = zero8();
#pragma unroll
        for (int ds = 0; ds < DIM / 32; ++ds) {
            Frag a, b;
            // A layout: lane row m=ln; elems e -> K = e + 16h (contiguous 32B)
            const uint4* ap = (const uint4*)(Qs + ln * DIM + ds * 32 + 16 * half);
            a.q[0] = ap[0];
            a.q[1] = ap[1];
            // B layout: lane col n=ln; elems e -> K = e + 16h (contiguous 32B)
            const uint4* bp = (const uint4*)(brow + ds * 32 + 16 * half);
            b.q[0] = bp[0];
            b.q[1] = bp[1];
            s = __builtin_amdgcn_wmma_f32_16x16x32_bf16(
                false, a.v, false, b.v, (short)0, s, false, false);
        }

        // ---- exp (no max needed: |score| small), mask tail, row sums, P ----
#pragma unroll
        for (int r = 0; r < 8; ++r) {
            float p = (latn < LATENT) ? __expf(s[r]) : 0.0f;
            float red = p;
            red += __shfl_xor(red, 1, 32);
            red += __shfl_xor(red, 2, 32);
            red += __shfl_xor(red, 4, 32);
            red += __shfl_xor(red, 8, 32);  // sum over 16 cols within lane-half
            rsum[r] += red;
            Ps[(r + 8 * half) * 128 + w * 16 + ln] = f2bf(p);
        }
        __syncthreads();  // P tile complete for all waves

        // ---- GEMM2: acc(16x96) += P(16x128) x V(128x96-slice) ----
#pragma unroll
        for (int kk = 0; kk < 4; ++kk) {
            Frag a;
            const uint4* pp = (const uint4*)(Ps + ln * 128 + kk * 32 + 16 * half);
            a.q[0] = pp[0];
            a.q[1] = pp[1];
#pragma unroll
            for (int nt = 0; nt < 6; ++nt) {
                const int dcol = w * 96 + nt * 16 + ln;
                Frag b;  // latT[dcol][lat]: contraction over latents is contiguous
                const uint4* bp = (const uint4*)(latT + (size_t)dcol * LATPAD +
                                                 kb * 128 + kk * 32 + 16 * half);
                b.q[0] = bp[0];
                b.q[1] = bp[1];
                acc[nt] = __builtin_amdgcn_wmma_f32_16x16x32_bf16(
                    false, a.v, false, b.v, (short)0, acc[nt], false, false);
            }
        }
        __syncthreads();  // protect Ps before next overwrite
    }

    // ---- combine per-wave row sums (ds_add_f32) ----
    if (lane == 0 || lane == 16) {
#pragma unroll
        for (int r = 0; r < 8; ++r) atomicAdd(&sumRow[r + 8 * half], rsum[r]);
    }
    __syncthreads();

    // ---- epilogue: sde = lang + P@V / sumexp; special-token override ----
#pragma unroll
    for (int r = 0; r < 8; ++r) {
        const int m    = r + 8 * half;
        const int grow = row0 + m;
        const float inv = 1.0f / sumRow[m];
        const int first = x[grow * CCH];
#pragma unroll
        for (int nt = 0; nt < 6; ++nt) {
            const int col = w * 96 + nt * 16 + ln;
            const int idx = grow * DIM + col;
            float v;
            if (first < 4)  // PAD/CLS/SEP/UNK
                v = emb[first * DIM + col];
            else
                v = out[idx] + acc[nt][r] * inv;  // out holds lang_emb f32
            out[idx] = v;
        }
    }
}

// ---------------------------------------------------------------------------
extern "C" void kernel_launch(void* const* d_in, const int* in_sizes, int n_in,
                              void* d_out, int out_size, void* d_ws, size_t ws_size,
                              hipStream_t stream) {
    const int*   x   = (const int*)d_in[0];
    const float* emb = (const float*)d_in[1];
    const float* lm  = (const float*)d_in[2];
    float* out = (float*)d_out;

    char* ws = (char*)d_ws;
    unsigned short* latbf  = (unsigned short*)(ws);
    unsigned short* latT   = (unsigned short*)(ws + (size_t)LATPAD * DIM * 2);
    unsigned short* langbf = (unsigned short*)(ws + (size_t)LATPAD * DIM * 4);
    // ws usage: 2 * 7.86 MB (latent bf16 + transpose) + 25.2 MB (lang bf16)

    lang_kernel<<<NROWS, 256, 0, stream>>>(x, emb, out, langbf);

    dim3 gb(LATPAD / 32, DIM / 32), bb(32, 8);
    prep_latent<<<gb, bb, 0, stream>>>(lm, latbf, latT);

    sde_attn<<<NROWS / 16, 256, 0, stream>>>(emb, x, langbf, latbf, latT, out);
}
